// INR_88158498718323
// MI455X (gfx1250) — matrix operand order repaired and checked by reading, not compiled, not run on hardware
//
#include <hip/hip_runtime.h>

typedef __attribute__((ext_vector_type(16))) _Float16 v16h;
typedef __attribute__((ext_vector_type(8)))  _Float16 v8h;
typedef __attribute__((ext_vector_type(8)))  float    v8f;
typedef __attribute__((ext_vector_type(2)))  int      v2i;

// ---------------------------------------------------------------------------
// f32 -> f16 elementwise conversion
// ---------------------------------------------------------------------------
__global__ __launch_bounds__(256) void cvt_f32_f16(const float* __restrict__ src,
                                                   _Float16* __restrict__ dst, int n) {
  int i = blockIdx.x * 256 + threadIdx.x;
  if (i < n) dst[i] = (_Float16)src[i];
}

// ---------------------------------------------------------------------------
// Zero the sentinel row (index npts) of each f16 feature buffer; gathers with
// nbr == -1 are redirected there so inner-loop loads are unconditional.
// ---------------------------------------------------------------------------
__global__ void zero_rows(_Float16* xh, _Float16* t0h, _Float16* t1h,
                          _Float16* t2h, int npts) {
  int i = threadIdx.x;
  if (i < 64) xh[(size_t)npts * 64 + i] = (_Float16)0.f;
  if (i < 32) {
    t0h[(size_t)npts * 32 + i] = (_Float16)0.f;
    t1h[(size_t)npts * 32 + i] = (_Float16)0.f;
    t2h[(size_t)npts * 32 + i] = (_Float16)0.f;
  }
}

// ---------------------------------------------------------------------------
// Repack [O, Cin, 32] f32 weights into WMMA B-fragment-ordered f16.
// Fragment f = (o*KC + kc)*2 + nt, 512 halfs per fragment, lane L owns 16
// contiguous halfs: half kk = W[o, kc*32 + 8*(L>>4) + (kk&7) + 16*(kk>>3), nt*16 + (L&15)].
// ---------------------------------------------------------------------------
__global__ __launch_bounds__(256) void prep_w(const float* __restrict__ w,
                                              _Float16* __restrict__ dst,
                                              int O, int KC, int Cin) {
  int d = blockIdx.x * 256 + threadIdx.x;
  int total = O * KC * 2 * 512;
  if (d >= total) return;
  int f = d >> 9, r = d & 511;
  int lane = r >> 4, kk = r & 15;
  int nt = f & 1, kcf = f >> 1;
  int o = kcf / KC, kc = kcf - o * KC;
  int ci = kc * 32 + ((lane >> 4) << 3) + (kk & 7) + ((kk >> 3) << 4);
  int co = nt * 16 + (lane & 15);
  dst[d] = (_Float16)w[((size_t)(o * Cin + ci)) * 32 + co];
}

// ---------------------------------------------------------------------------
// Tiled (sparse) conv via WMMA. One wave = two 16-point M-tiles (32 points):
// each B fragment feeds two WMMAs. LDS holds *precomputed clamped byte
// offsets* (tile0/tile1 interleaved -> one ds_load_b64 per kernel offset);
// absent voxels point at the zeroed sentinel row. 32-bit offsets + uniform
// base let the backend use saddr-form global loads with immediate offsets.
//   CIN : 32 or 64 input channels (f16), Cout = 32 (two 16-wide N tiles)
//   O   : 27 (sparse 3x3x3, nbr map) or 1 (dense 1x1)
//   MODE 0: relu -> f16 intermediate [npts+1, 32]
//   MODE 1: f32 out + residual: d_out[:, COLOFF..+31] = acc + feats[:, COLOFF..]
// ---------------------------------------------------------------------------
template<int CIN, int O, int MODE, int COLOFF>
__global__ __launch_bounds__(256) void spconv(const _Float16* __restrict__ xin,
                                              const _Float16* __restrict__ wfrag,
                                              const int* __restrict__ nbr,
                                              const float* __restrict__ feats,
                                              void* __restrict__ outp, int npts) {
  constexpr int KC = CIN / 32;
  constexpr int WHALFS = O * KC * 2 * 512;
  constexpr int RB = CIN * 2;               // row stride in bytes (f16)
  __shared__ alignas(32) _Float16 wlds[WHALFS];
  __shared__ alignas(8) int nlds[(O == 27) ? (8 * 2 * 432) : 8];

  for (int i = threadIdx.x; i < WHALFS / 8; i += 256)
    ((v8h*)wlds)[i] = ((const v8h*)wfrag)[i];
  __syncthreads();

  const int wave  = threadIdx.x >> 5;
  const int lane  = threadIdx.x & 31;
  const int tile0 = (blockIdx.x * 8 + wave) * 32;
  const int tile1 = tile0 + 16;
  if (tile0 >= npts) return;
  const bool has1 = (tile1 < npts);

  if constexpr (O == 27) {
    // Interleave tile0/tile1 byte offsets so the hot loop reads both with
    // one ds_load_b64. Clamp to the sentinel row here, once per entry.
    for (int j = lane; j < 432; j += 32) {
      int q0 = nbr[(size_t)tile0 * 27 + j];
      int q1 = has1 ? nbr[(size_t)tile1 * 27 + j] : -1;
      nlds[wave * 864 + 2 * j]     = (q0 < 0 ? npts : q0) * RB;
      nlds[wave * 864 + 2 * j + 1] = (q1 < 0 ? npts : q1) * RB;
    }
  }

  const int bkb = (lane >> 4) * 16;  // A-fragment K byte base inside a 32-chunk
  const int n   = lane & 15;         // point-in-tile (A row) == out column id
  const char* xb = (const char*)xin;
  v8f c00 = {}, c01 = {}, c10 = {}, c11 = {};

  for (int o = 0; o < O; ++o) {
    unsigned off0, off1;
    if constexpr (O == 27) {
      v2i offs = *(const v2i*)&nlds[wave * 864 + (n * 27 + o) * 2];
      off0 = (unsigned)(offs.x + bkb);
      off1 = (unsigned)(offs.y + bkb);
    } else {
      off0 = (unsigned)((tile0 + n) * RB + bkb);
      off1 = (unsigned)((has1 ? (tile1 + n) : npts) * RB + bkb);
    }
#pragma unroll
    for (int kc = 0; kc < KC; ++kc) {
      union { v16h v; v8h h[2]; } a0, a1;
      a0.h[0] = *(const v8h*)(xb + off0 + kc * 64);
      a0.h[1] = *(const v8h*)(xb + off0 + kc * 64 + 32);
      a1.h[0] = *(const v8h*)(xb + off1 + kc * 64);
      a1.h[1] = *(const v8h*)(xb + off1 + kc * 64 + 32);

      const _Float16* wb = &wlds[((o * KC + kc) * 2) * 512 + lane * 16];
      v16h b0 = *(const v16h*)wb;          // N tile 0
      v16h b1 = *(const v16h*)(wb + 512);  // N tile 1

      c00 = __builtin_amdgcn_wmma_f32_16x16x32_f16(false, a0.v, false, b0,
                                                   (short)0, c00, false, false);
      c01 = __builtin_amdgcn_wmma_f32_16x16x32_f16(false, a0.v, false, b1,
                                                   (short)0, c01, false, false);
      c10 = __builtin_amdgcn_wmma_f32_16x16x32_f16(false, a1.v, false, b0,
                                                   (short)0, c10, false, false);
      c11 = __builtin_amdgcn_wmma_f32_16x16x32_f16(false, a1.v, false, b1,
                                                   (short)0, c11, false, false);
    }
  }

  // C/D layout: VGPR v -> M = v + 8*(lane>>4), N = lane&15
  const int hb = (lane >> 4) << 3;
  if constexpr (MODE == 0) {
    _Float16* out = (_Float16*)outp;
#pragma unroll
    for (int v = 0; v < 8; ++v) {
      size_t r0 = (size_t)(tile0 + hb + v) * 32;
      out[r0 + n]      = (_Float16)fmaxf(c00[v], 0.f);
      out[r0 + 16 + n] = (_Float16)fmaxf(c01[v], 0.f);
    }
    if (has1) {
#pragma unroll
      for (int v = 0; v < 8; ++v) {
        size_t r1 = (size_t)(tile1 + hb + v) * 32;
        out[r1 + n]      = (_Float16)fmaxf(c10[v], 0.f);
        out[r1 + 16 + n] = (_Float16)fmaxf(c11[v], 0.f);
      }
    }
  } else {
    float* out = (float*)outp;
#pragma unroll
    for (int v = 0; v < 8; ++v) {
      size_t r0 = (size_t)(tile0 + hb + v) * 64 + COLOFF;
      out[r0 + n]      = c00[v] + feats[r0 + n];
      out[r0 + 16 + n] = c01[v] + feats[r0 + 16 + n];
    }
    if (has1) {
#pragma unroll
      for (int v = 0; v < 8; ++v) {
        size_t r1 = (size_t)(tile1 + hb + v) * 64 + COLOFF;
        out[r1 + n]      = c10[v] + feats[r1 + n];
        out[r1 + 16 + n] = c11[v] + feats[r1 + 16 + n];
      }
    }
  }
}

// ---------------------------------------------------------------------------
extern "C" void kernel_launch(void* const* d_in, const int* in_sizes, int n_in,
                              void* d_out, int out_size, void* d_ws, size_t ws_size,
                              hipStream_t stream) {
  const float* feats = (const float*)d_in[0];
  const float* w00   = (const float*)d_in[1];   // [27,64,32]
  const float* w01   = (const float*)d_in[2];   // [27,32,32]
  const float* w10   = (const float*)d_in[3];   // [64,32]
  const float* w11   = (const float*)d_in[4];   // [27,32,32]
  const float* w12   = (const float*)d_in[5];   // [32,32]
  const int*   nbr   = (const int*)d_in[6];     // [N,27]
  float* out = (float*)d_out;

  const int NPTS = in_sizes[0] / 64;            // 250000
  // +1 sentinel (zero) row per buffer; keep offsets 256B-aligned
  size_t szX = ((size_t)(NPTS + 1) * 64 * sizeof(_Float16) + 255) & ~(size_t)255;
  size_t szH = ((size_t)(NPTS + 1) * 32 * sizeof(_Float16) + 255) & ~(size_t)255;

  char* ws = (char*)d_ws;
  _Float16* xh   = (_Float16*)ws;
  _Float16* t0h  = (_Float16*)(ws + szX);
  _Float16* t1h  = (_Float16*)(ws + szX + szH);
  _Float16* t2h  = (_Float16*)(ws + szX + 2 * szH);
  char* wbase = ws + szX + 3 * szH;
  _Float16* w00h = (_Float16*)(wbase);                                   // 110592 B
  _Float16* w01h = (_Float16*)(wbase + 110592);                          //  55296 B
  _Float16* w10h = (_Float16*)(wbase + 110592 + 55296);                  //   4096 B
  _Float16* w11h = (_Float16*)(wbase + 110592 + 55296 + 4096);           //  55296 B
  _Float16* w12h = (_Float16*)(wbase + 110592 + 55296 + 4096 + 55296);   //   2048 B

  int nE = NPTS * 64;
  cvt_f32_f16<<<(nE + 255) / 256, 256, 0, stream>>>(feats, xh, nE);
  zero_rows<<<1, 64, 0, stream>>>(xh, t0h, t1h, t2h, NPTS);
  prep_w<<<(55296 + 255) / 256, 256, 0, stream>>>(w00, w00h, 27, 2, 64);
  prep_w<<<(27648 + 255) / 256, 256, 0, stream>>>(w01, w01h, 27, 1, 32);
  prep_w<<<(2048  + 255) / 256, 256, 0, stream>>>(w10, w10h, 1, 2, 64);
  prep_w<<<(27648 + 255) / 256, 256, 0, stream>>>(w11, w11h, 27, 1, 32);
  prep_w<<<(1024  + 255) / 256, 256, 0, stream>>>(w12, w12h, 1, 1, 32);

  int tiles  = (NPTS + 15) / 16;
  int pairs  = (tiles + 1) / 2;
  int blocks = (pairs + 7) / 8;
  // branch 0: conv3(64->32)+relu ; conv3(32->32) + feats[:, :32] residual
  spconv<64, 27, 0, 0 ><<<blocks, 256, 0, stream>>>(xh,  w00h, nbr,     nullptr, t0h,  NPTS);
  spconv<32, 27, 1, 0 ><<<blocks, 256, 0, stream>>>(t0h, w01h, nbr,     feats,   out,  NPTS);
  // branch 1: 1x1(64->32)+relu ; conv3(32->32)+relu ; 1x1(32->32) + feats[:,32:] residual
  spconv<64, 1,  0, 0 ><<<blocks, 256, 0, stream>>>(xh,  w10h, nullptr, nullptr, t1h,  NPTS);
  spconv<32, 27, 0, 0 ><<<blocks, 256, 0, stream>>>(t1h, w11h, nbr,     nullptr, t2h,  NPTS);
  spconv<32, 1,  1, 32><<<blocks, 256, 0, stream>>>(t2h, w12h, nullptr, feats,   out,  NPTS);
}
